// ClassifyMLPHeadForKCRWithConcatChoices_66898410603080
// MI455X (gfx1250) — matrix operand order repaired and checked by reading, not compiled, not run on hardware
//
#include <hip/hip_runtime.h>
#include <hip/hip_bf16.h>

typedef __attribute__((ext_vector_type(16))) __bf16          v16bf;
typedef __attribute__((ext_vector_type(8)))  float           v8f;
typedef __attribute__((ext_vector_type(8)))  unsigned short  v8u;
typedef __attribute__((ext_vector_type(16))) unsigned short  v16u;

union Frag { v8u h[2]; v16bf b; v16u u; };

#define BB 16
#define LL 2048
#define HH 1024
#define AA 1024
#define CC 5
#define MBLK 32        // rows of (B*L) per block: two 16-row WMMA tiles share each B fragment
#define XSTRIDE 1032   // padded LDS row stride (bf16 elems): 516 dwords % 64 banks = 4 -> conflict-free

__device__ __forceinline__ unsigned short f2bf(float f) {
    union { float f; unsigned u; } x; x.f = f;
    unsigned u = x.u;
    unsigned r = (u + 0x7FFFu + ((u >> 16) & 1u)) >> 16;   // round-to-nearest-even
    return (unsigned short)r;
}

// ---------------- W_hidden fp32 -> bf16 (done once; result stays L2-resident) --------------
__global__ void __launch_bounds__(256) convert_w_kernel(const float* __restrict__ W,
                                                        unsigned short* __restrict__ Wbf) {
    int i = blockIdx.x * 256 + threadIdx.x;            // float4 granularity, 1024 blocks
    float4 f = ((const float4*)W)[i];
    uint2 o;
    o.x = (unsigned)f2bf(f.x) | ((unsigned)f2bf(f.y) << 16);
    o.y = (unsigned)f2bf(f.z) | ((unsigned)f2bf(f.w) << 16);
    ((uint2*)Wbf)[i] = o;
}

// ---------------- r[b,l] = <X[b,l,:], W_cls> (memory-bound GEMV, wave per row) --------------
__global__ void __launch_bounds__(256) gemv_r_kernel(const float* __restrict__ X,
                                                     const float* __restrict__ Wcls,
                                                     float* __restrict__ r) {
    int row  = blockIdx.x * 8 + (threadIdx.x >> 5);
    int lane = threadIdx.x & 31;
    const float4* xr = (const float4*)(X + (size_t)row * HH);
    const float4* w4 = (const float4*)Wcls;
    float s = 0.f;
    #pragma unroll
    for (int i = lane; i < HH / 4; i += 32) {
        float4 a = xr[i], b = w4[i];
        s += a.x * b.x + a.y * b.y + a.z * b.z + a.w * b.w;
    }
    #pragma unroll
    for (int off = 16; off > 0; off >>= 1) s += __shfl_xor(s, off, 32);
    if (lane == 0) r[row] = s;
}

// ---------------- fused keys-GEMM + tanh + query-dot  ->  probs[b*L+l] ----------------------
// One block = 32 rows of (B*L) (two 16-row WMMA tiles). 8 waves x 8 a-tiles = all 64 tiles
// of A=1024. Both row tiles share every B fragment -> halves W_hidden L2 traffic.
__global__ void __launch_bounds__(256) wmma_probs_kernel(const float* __restrict__ X,
                                                         const unsigned short* __restrict__ Wbf,
                                                         const float* __restrict__ query,
                                                         const float* __restrict__ bhid,
                                                         float* __restrict__ probs) {
    __shared__ unsigned short sx[MBLK * XSTRIDE];   // 66 KB bf16 X tile
    __shared__ float red[MBLK];

    const int rt = blockIdx.x;
    const float* xb = X + (size_t)rt * MBLK * HH;

    // stage X tile -> LDS as bf16 (coalesced float4 reads)
    for (int i = threadIdx.x; i < MBLK * HH / 4; i += 256) {
        float4 f = ((const float4*)xb)[i];
        int e = i * 4, row = e >> 10, col = e & (HH - 1);
        uint2 o;
        o.x = (unsigned)f2bf(f.x) | ((unsigned)f2bf(f.y) << 16);
        o.y = (unsigned)f2bf(f.z) | ((unsigned)f2bf(f.w) << 16);
        *(uint2*)&sx[row * XSTRIDE + col] = o;
    }
    if (threadIdx.x < MBLK) red[threadIdx.x] = 0.f;
    __syncthreads();

    const int wave = threadIdx.x >> 5;
    const int lane = threadIdx.x & 31;
    const int n    = lane & 15;        // A-row (M) for A-frags / column (N) for B-frag
    const int hi   = lane >> 4;
    const int kbA  = hi * 8;           // 16-bit A VGPR layout: K pairs 0..7 / 8..15 per half
    const int kbB  = hi * 8;           // 16-bit B layout: K pairs 0..7 / 8..15 per half (same as A)

    const unsigned short* xrow0 = &sx[n * XSTRIDE];          // row tile 0: rows 0..15
    const unsigned short* xrow1 = &sx[(16 + n) * XSTRIDE];   // row tile 1: rows 16..31
    float s0[8] = {0.f, 0.f, 0.f, 0.f, 0.f, 0.f, 0.f, 0.f};
    float s1[8] = {0.f, 0.f, 0.f, 0.f, 0.f, 0.f, 0.f, 0.f};

    for (int t = 0; t < 8; ++t) {
        const int a0 = (wave * 8 + t) * 16;
        const unsigned short* wrow = Wbf + (size_t)(a0 + n) * HH;
        if (t < 7) __builtin_prefetch(wrow + (size_t)16 * HH, 0, 1);  // next a-tile's W row

        v8f acc0 = {}, acc1 = {};
        Frag bcur;                                    // B double-buffer: prefetch k+1
        bcur.h[0] = *(const v8u*)(wrow + kbB);
        bcur.h[1] = *(const v8u*)(wrow + kbB + 16);
        #pragma unroll 4
        for (int ks = 0; ks < HH / 32; ++ks) {
            const int k0 = ks * 32;
            Frag bnxt = bcur;
            if (ks < HH / 32 - 1) {
                bnxt.h[0] = *(const v8u*)(wrow + k0 + 32 + kbB);
                bnxt.h[1] = *(const v8u*)(wrow + k0 + 32 + kbB + 16);
            }
            Frag a0f, a1f;
            a0f.h[0] = *(const v8u*)(xrow0 + k0 + kbA);
            a0f.h[1] = *(const v8u*)(xrow0 + k0 + 16 + kbA);
            a1f.h[0] = *(const v8u*)(xrow1 + k0 + kbA);
            a1f.h[1] = *(const v8u*)(xrow1 + k0 + 16 + kbA);
            acc0 = __builtin_amdgcn_wmma_f32_16x16x32_bf16(false, a0f.b, false, bcur.b,
                                                           (short)0, acc0, false, false);
            acc1 = __builtin_amdgcn_wmma_f32_16x16x32_bf16(false, a1f.b, false, bcur.b,
                                                           (short)0, acc1, false, false);
            bcur = bnxt;
        }
        const float q  = query[a0 + n];
        const float bb = bhid[a0 + n];
        #pragma unroll
        for (int v = 0; v < 8; ++v) {
            s0[v] += tanhf(acc0[v] + bb) * q;
            s1[v] += tanhf(acc1[v] + bb) * q;
        }
    }
    // reduce row partials across the 16 lanes of each half (C/D layout: M = v + 8*hi)
    #pragma unroll
    for (int v = 0; v < 8; ++v) {
        float a = s0[v], b = s1[v];
        a += __shfl_xor(a, 1, 32);  b += __shfl_xor(b, 1, 32);
        a += __shfl_xor(a, 2, 32);  b += __shfl_xor(b, 2, 32);
        a += __shfl_xor(a, 4, 32);  b += __shfl_xor(b, 4, 32);
        a += __shfl_xor(a, 8, 32);  b += __shfl_xor(b, 8, 32);
        s0[v] = a; s1[v] = b;
    }
    if (n == 0) {
        #pragma unroll
        for (int v = 0; v < 8; ++v) {
            atomicAdd(&red[hi * 8 + v], s0[v]);
            atomicAdd(&red[16 + hi * 8 + v], s1[v]);
        }
    }
    __syncthreads();
    if (threadIdx.x < MBLK) probs[rt * MBLK + threadIdx.x] = red[threadIdx.x];
}

// ---------------- per-batch: qvar, softmax(probs*scale*mask), segment sums ------------------
__device__ __forceinline__ float blockSum(float v, float* sbuf) {
    #pragma unroll
    for (int off = 16; off > 0; off >>= 1) v += __shfl_xor(v, off, 32);
    __syncthreads();
    if ((threadIdx.x & 31) == 0) sbuf[threadIdx.x >> 5] = v;
    __syncthreads();
    float tot = 0.f;
    #pragma unroll
    for (int i = 0; i < 8; ++i) tot += sbuf[i];
    return tot;
}
__device__ __forceinline__ float blockMax(float v, float* sbuf) {
    #pragma unroll
    for (int off = 16; off > 0; off >>= 1) v = fmaxf(v, __shfl_xor(v, off, 32));
    __syncthreads();
    if ((threadIdx.x & 31) == 0) sbuf[threadIdx.x >> 5] = v;
    __syncthreads();
    float m = sbuf[0];
    #pragma unroll
    for (int i = 1; i < 8; ++i) m = fmaxf(m, sbuf[i]);
    return m;
}

__global__ void __launch_bounds__(256) final_kernel(const int* __restrict__ attn,
                                                    const int* __restrict__ mlm,
                                                    const float* __restrict__ query,
                                                    const float* __restrict__ Wcls,
                                                    const float* __restrict__ bcls,
                                                    const float* __restrict__ probs,
                                                    const float* __restrict__ rbuf,
                                                    float* __restrict__ out) {
    __shared__ float att[LL];
    __shared__ float sbuf[8];
    __shared__ int   starts[CC];
    __shared__ int   sEnd;
    __shared__ float sScale, sWsum;
    const int b = blockIdx.x, t = threadIdx.x;

    float ssum = 0.f, ssq = 0.f, ws = 0.f;
    for (int i = t; i < AA; i += 256) {
        float qv = query[i];
        ssum += qv; ssq += qv * qv; ws += Wcls[i];
    }
    ssum = blockSum(ssum, sbuf);
    ssq  = blockSum(ssq, sbuf);
    ws   = blockSum(ws, sbuf);
    float endf = 0.f;
    for (int i = t; i < LL; i += 256) endf += (float)attn[b * LL + i];
    endf = blockSum(endf, sbuf);
    if (t == 0) {
        sEnd = (int)(endf + 0.5f);
        float var = (ssq - ssum * ssum / (float)AA) / (float)(AA - 1);  // ddof=1
        sScale = rsqrtf((float)AA * var);
        sWsum = ws;
        int c = 0;
        for (int l = 0; l < LL && c < CC; ++l)
            if (mlm[b * LL + l] > 0) starts[c++] = l;
    }
    __syncthreads();

    // faithful logits: probs_scaled * mask, mask = (1-attn)*(-1000)
    float mx = -3.402823e38f;
    for (int i = t; i < LL; i += 256) {
        float m  = attn[b * LL + i] ? 0.f : -1000.f;
        float lg = probs[b * LL + i] * sScale * m;
        att[i] = lg;
        mx = fmaxf(mx, lg);
    }
    mx = blockMax(mx, sbuf);
    float es = 0.f;
    for (int i = t; i < LL; i += 256) {
        float e = __expf(att[i] - mx);
        att[i] = e;
        es += e;
    }
    es = blockSum(es, sbuf);
    const float inv = 1.f / es;

    // logits[b,c] = b_cls + sum_{l in [start+1, bound)} ( att*Wsum + r[b,l] )
    for (int c = 0; c < CC; ++c) {
        int st = starts[c] + 1;
        int bd = (c < CC - 1) ? (starts[c + 1] - 1) : (sEnd - 1);
        float s = 0.f;
        for (int l = st + t; l < bd; l += 256)
            s += att[l] * inv * sWsum + rbuf[b * LL + l];
        s = blockSum(s, sbuf);
        if (t == 0) out[b * CC + c] = s + bcls[0];
    }
}

extern "C" void kernel_launch(void* const* d_in, const int* in_sizes, int n_in,
                              void* d_out, int out_size, void* d_ws, size_t ws_size,
                              hipStream_t stream) {
    const float* X    = (const float*)d_in[0];
    const int*   attn = (const int*)d_in[1];
    const int*   mlm  = (const int*)d_in[2];
    const float* Wh   = (const float*)d_in[3];
    const float* bh   = (const float*)d_in[4];
    const float* q    = (const float*)d_in[5];
    const float* Wc   = (const float*)d_in[6];
    const float* bc   = (const float*)d_in[7];
    float* out = (float*)d_out;

    char* ws = (char*)d_ws;
    unsigned short* Wbf  = (unsigned short*)ws;                       // 2 MB bf16 W_hidden
    float* rbuf  = (float*)(ws + (size_t)2 * 1024 * 1024);            // 128 KB r[b,l]
    float* probs = (float*)(ws + (size_t)2 * 1024 * 1024 + 131072);   // 128 KB probs[b,l]

    convert_w_kernel<<<(AA * HH / 4) / 256, 256, 0, stream>>>(Wh, Wbf);
    gemv_r_kernel<<<(BB * LL) / 8, 256, 0, stream>>>(X, Wc, rbuf);
    wmma_probs_kernel<<<(BB * LL) / MBLK, 256, 0, stream>>>(X, Wbf, q, bh, probs);
    final_kernel<<<BB, 256, 0, stream>>>(attn, mlm, q, Wc, bc, probs, rbuf, out);
}
